// Step3TextDecoderLayer_953482740197
// MI455X (gfx1250) — compile-verified, matrix operand order checked by reading, not run on hardware
//
#include <hip/hip_runtime.h>

// ---------------------------------------------------------------------------
// Types / helpers
// ---------------------------------------------------------------------------
typedef __attribute__((ext_vector_type(16))) __bf16 v16bf;
typedef __attribute__((ext_vector_type(8)))  float  v8f;

#define TOK   2048   // B*S
#define HID   4096
#define QS_   2048
#define DH_   128
#define NH_   32
#define E_    8
#define MI_   1024
#define NQKV  2304   // QS + 2*DH

__device__ __forceinline__ unsigned short f2bf(float x) {
  unsigned int u = __float_as_uint(x);
  unsigned int r = (u + 0x7FFFu + ((u >> 16) & 1u)) >> 16;
  return (unsigned short)r;
}
__device__ __forceinline__ float bf2f(unsigned short h) {
  return __uint_as_float(((unsigned int)h) << 16);
}
__device__ __forceinline__ unsigned int pack_bf2(float lo, float hi) {
  return (unsigned int)f2bf(lo) | ((unsigned int)f2bf(hi) << 16);
}

// DPP cross-lane moves within a 16-lane row (no DS unit, no waits).
template <int CTRL>
__device__ __forceinline__ float dpp_mov(float x) {
  return __int_as_float(
      __builtin_amdgcn_update_dpp(0, __float_as_int(x), CTRL, 0xF, 0xF, true));
}
// Butterfly reductions over the 16-lane half-wave (both halves independent).
__device__ __forceinline__ float red_max16(float x) {
  x = fmaxf(x, dpp_mov<0xB1>(x));   // quad_perm(1,0,3,2)  : xor 1
  x = fmaxf(x, dpp_mov<0x4E>(x));   // quad_perm(2,3,0,1)  : xor 2
  x = fmaxf(x, dpp_mov<0x141>(x));  // row_half_mirror     : xor 4 (quads uniform)
  x = fmaxf(x, dpp_mov<0x140>(x));  // row_mirror          : xor 8
  return x;
}
__device__ __forceinline__ float red_sum16(float x) {
  x += dpp_mov<0xB1>(x);
  x += dpp_mov<0x4E>(x);
  x += dpp_mov<0x141>(x);
  x += dpp_mov<0x140>(x);
  return x;
}

union FragU { v16bf v; unsigned short u[16]; };

// Load a 16xK=32 A-style (or mirrored B-style) WMMA fragment from LDS.
// Layout (CDNA5 ISA 7.12.2): lane l: row = row0 + (l&15), kbase = (l>>4)*8,
// halves[0..7] = [kbase..kbase+7], halves[8..15] = [kbase+16..kbase+23].
__device__ __forceinline__ v16bf load_frag(const unsigned short* base, int stride,
                                           int row0, int lane) {
  const unsigned short* p = base + (row0 + (lane & 15)) * stride + ((lane >> 4) << 3);
  FragU f;
#pragma unroll
  for (int j = 0; j < 8; ++j) { f.u[j] = p[j]; f.u[8 + j] = p[16 + j]; }
  return f.v;
}

__device__ __forceinline__ v8f wmma_bf16(v16bf a, v16bf b, v8f c) {
  return __builtin_amdgcn_wmma_f32_16x16x32_bf16(false, a, false, b, (short)0, c,
                                                 false, false);
}

// ---------------------------------------------------------------------------
// Fused add + RMSNorm: x = a (+ b); res_out = x; h_out = bf16(x * rsqrt(ms) * w)
// ---------------------------------------------------------------------------
__global__ __launch_bounds__(256) void add_rmsnorm_kernel(
    const float* __restrict__ a, int lda, const float* __restrict__ b,
    const float* __restrict__ w, float* __restrict__ res_out,
    unsigned short* __restrict__ h_out, int cols) {
  int row = blockIdx.x, t = threadIdx.x;
  const float* pa = a + (size_t)row * lda;
  const float* pb = b ? b + (size_t)row * cols : nullptr;
  int n = cols >> 8;
  float xv[16];
  float ss = 0.f;
  for (int j = 0; j < n; ++j) {
    int i = t + (j << 8);
    float x = pa[i];
    if (pb) x += pb[i];
    xv[j] = x;
    ss += x * x;
  }
  for (int m = 16; m >= 1; m >>= 1) ss += __shfl_xor(ss, m, 32);
  __shared__ float sred[8];
  if ((t & 31) == 0) sred[t >> 5] = ss;
  __syncthreads();
  float tot = 0.f;
#pragma unroll
  for (int k = 0; k < 8; ++k) tot += sred[k];
  float r = rsqrtf(tot / (float)cols + 1e-5f);
  for (int j = 0; j < n; ++j) {
    int i = t + (j << 8);
    h_out[(size_t)row * cols + i] = f2bf(xv[j] * r * w[i]);
    if (res_out) res_out[(size_t)row * cols + i] = xv[j];
  }
}

// ---------------------------------------------------------------------------
// Generic tiled WMMA GEMM: C[M,N] = A(bf16)[M,K] * B(fp32->bf16)[K,N]
// Block tile 128x128, BK=32, 8 waves each owning 32x64.
// Register-staged software pipeline: next tile's global loads overlap WMMA.
// ---------------------------------------------------------------------------
struct GemmP {
  const unsigned short* A;   // bf16 activations
  const float* B;            // fp32 weights (row-major K x N)
  int M, N, K, lda, ldc;
  float* out_f32;
  unsigned short* out_b16;
  const float* res;          // EMODE 2: residual added (stride ldc)
  const int* rowidx;         // AMODE 1
  const int* cnt;            // AMODE 1/2
  const int* tok;            // EMODE 3
  const int* krank;          // EMODE 3
  const float* wt;           // EMODE 3
  float* out2;               // EMODE 3 (write) / 4 (read)
};

// AMODE: 0 direct, 1 indirect rows via rowidx (count-bounded), 2 direct count-bounded
// EMODE: 0 f32 store, 1 bf16 store, 2 acc+res -> f32, 3 weighted scatter to out2,
//        4 acc + out2[t][0] + out2[t][1] -> f32
template <int AMODE, int EMODE>
__global__ __launch_bounds__(256) void gemm_bf16_kernel(GemmP p) {
  __shared__ unsigned short sA[128 * 36];
  __shared__ unsigned short sB[128 * 36];
  int t = threadIdx.x, lane = t & 31, wid = t >> 5;
  int bn0 = blockIdx.x * 128, bm0 = blockIdx.y * 128;
  int count = p.M;
  if (AMODE != 0) {
    count = *p.cnt;
    if (bm0 >= count) return;
  }
  int wm = wid & 3, wn = wid >> 2;

  v8f acc[2][4];
#pragma unroll
  for (int i = 0; i < 2; ++i)
#pragma unroll
    for (int j = 0; j < 4; ++j)
#pragma unroll
      for (int r = 0; r < 8; ++r) acc[i][j][r] = 0.f;

  int arow = t >> 1, ac0 = (t & 1) << 4;     // A: 128 rows x 32 halves per tile
  int bk2 = (t >> 4) << 1, bc8 = (t & 15) << 3;  // B: rows bk2,bk2+1, cols bc8..+7
  int nkt = p.K >> 5;

  uint4 ra0, ra1;
  float4 rb0, rb1, rb2, rb3;

  auto loadA = [&](int kt) {
    ra0 = make_uint4(0, 0, 0, 0);
    ra1 = ra0;
    int rg = bm0 + arow;
    bool valid = true;
    const unsigned short* src = nullptr;
    if (AMODE == 0) {
      src = p.A + (size_t)rg * p.lda + kt * 32 + ac0;
    } else if (AMODE == 1) {
      valid = rg < count;
      if (valid) src = p.A + (size_t)p.rowidx[rg] * p.lda + kt * 32 + ac0;
    } else {
      valid = rg < count;
      if (valid) src = p.A + (size_t)rg * p.lda + kt * 32 + ac0;
    }
    if (valid) {
      const uint4* s4 = (const uint4*)src;
      ra0 = s4[0];
      ra1 = s4[1];
    }
  };
  auto loadB = [&](int kt) {
    const float* bs = p.B + (size_t)(kt * 32 + bk2) * p.N + bn0 + bc8;
    const float4* r0 = (const float4*)bs;
    const float4* r1 = (const float4*)(bs + p.N);
    rb0 = r0[0]; rb1 = r0[1];
    rb2 = r1[0]; rb3 = r1[1];
  };

  loadA(0);
  loadB(0);

  for (int kt = 0; kt < nkt; ++kt) {
    __syncthreads();
    // ---- commit staged registers to LDS ----
    {
      uint2* dst = (uint2*)(sA + arow * 36 + ac0);
      dst[0] = make_uint2(ra0.x, ra0.y);
      dst[1] = make_uint2(ra0.z, ra0.w);
      dst[2] = make_uint2(ra1.x, ra1.y);
      dst[3] = make_uint2(ra1.z, ra1.w);
      unsigned int w0 = pack_bf2(rb0.x, rb2.x), w1 = pack_bf2(rb0.y, rb2.y);
      unsigned int w2 = pack_bf2(rb0.z, rb2.z), w3 = pack_bf2(rb0.w, rb2.w);
      unsigned int w4 = pack_bf2(rb1.x, rb3.x), w5 = pack_bf2(rb1.y, rb3.y);
      unsigned int w6 = pack_bf2(rb1.z, rb3.z), w7 = pack_bf2(rb1.w, rb3.w);
      *(unsigned int*)(sB + (bc8 + 0) * 36 + bk2) = w0;
      *(unsigned int*)(sB + (bc8 + 1) * 36 + bk2) = w1;
      *(unsigned int*)(sB + (bc8 + 2) * 36 + bk2) = w2;
      *(unsigned int*)(sB + (bc8 + 3) * 36 + bk2) = w3;
      *(unsigned int*)(sB + (bc8 + 4) * 36 + bk2) = w4;
      *(unsigned int*)(sB + (bc8 + 5) * 36 + bk2) = w5;
      *(unsigned int*)(sB + (bc8 + 6) * 36 + bk2) = w6;
      *(unsigned int*)(sB + (bc8 + 7) * 36 + bk2) = w7;
    }
    __syncthreads();
    // ---- issue next tile's global loads (overlap with WMMA below) ----
    if (kt + 1 < nkt) {
      loadA(kt + 1);
      loadB(kt + 1);
    }
    if (kt + 3 < nkt) {
      __builtin_prefetch(p.B + (size_t)(kt * 32 + 96 + bk2) * p.N + bn0 + bc8, 0, 1);
    }
    // ---- compute ----
    v16bf fa[2];
#pragma unroll
    for (int mt = 0; mt < 2; ++mt) fa[mt] = load_frag(sA, 36, wm * 32 + mt * 16, lane);
#pragma unroll
    for (int nt = 0; nt < 4; ++nt) {
      v16bf fb = load_frag(sB, 36, wn * 64 + nt * 16, lane);
#pragma unroll
      for (int mt = 0; mt < 2; ++mt) acc[mt][nt] = wmma_bf16(fa[mt], fb, acc[mt][nt]);
    }
  }

  // ---- epilogue ----
  int hi = lane >> 4, lo = lane & 15;
#pragma unroll
  for (int mt = 0; mt < 2; ++mt)
#pragma unroll
    for (int nt = 0; nt < 4; ++nt)
#pragma unroll
      for (int r = 0; r < 8; ++r) {
        int rg = bm0 + wm * 32 + mt * 16 + r + 8 * hi;
        int cg = bn0 + wn * 64 + nt * 16 + lo;
        if (AMODE != 0 && rg >= count) continue;
        float v = acc[mt][nt][r];
        if (EMODE == 0) {
          p.out_f32[(size_t)rg * p.ldc + cg] = v;
        } else if (EMODE == 1) {
          p.out_b16[(size_t)rg * p.ldc + cg] = f2bf(v);
        } else if (EMODE == 2) {
          p.out_f32[(size_t)rg * p.ldc + cg] = v + p.res[(size_t)rg * p.ldc + cg];
        } else if (EMODE == 3) {
          int tk = p.tok[rg], kr = p.krank[rg];
          float w = p.wt[rg];
          p.out2[((size_t)tk * 2 + kr) * HID + cg] = w * v;
        } else {
          p.out_f32[(size_t)rg * HID + cg] =
              v + p.out2[((size_t)rg * 2 + 0) * HID + cg] +
              p.out2[((size_t)rg * 2 + 1) * HID + cg];
        }
      }
}

// ---------------------------------------------------------------------------
// RoPE + repack: qproj fp32 [T,4096] -> q_r bf16 [(b*32+h)*1024+s][128];
// k,v slices of qkv fp32 [T,2304] -> k_r/v_r bf16 [T][128]
// ---------------------------------------------------------------------------
__global__ __launch_bounds__(256) void rope_kernel(
    const long long* __restrict__ pos, const float* __restrict__ qproj,
    const float* __restrict__ qkv, unsigned short* __restrict__ q_r,
    unsigned short* __restrict__ k_r, unsigned short* __restrict__ v_r) {
  int row = blockIdx.x;             // token
  int b = row >> 10, s = row & 1023;
  float p = (float)pos[row];
  int t = threadIdx.x;
  const float L2T_64 = 0.20762050f; // log2(10000)/64
  for (int pr = t; pr < NH_ * 64; pr += 256) {
    int h = pr >> 6, j = pr & 63;
    float inv = exp2f(-(float)j * L2T_64);
    float ang = p * inv, c = __cosf(ang), sn = __sinf(ang);
    const float* q = qproj + (size_t)row * HID + h * DH_;
    float x1 = q[j], x2 = q[j + 64];
    size_t ob = ((size_t)(b * NH_ + h) * 1024 + s) * DH_;
    q_r[ob + j]      = f2bf(x1 * c - x2 * sn);
    q_r[ob + j + 64] = f2bf(x2 * c + x1 * sn);
  }
  if (t < 64) {
    int j = t;
    float inv = exp2f(-(float)j * L2T_64);
    float ang = p * inv, c = __cosf(ang), sn = __sinf(ang);
    const float* k = qkv + (size_t)row * NQKV + QS_;
    float x1 = k[j], x2 = k[j + 64];
    k_r[(size_t)row * DH_ + j]      = f2bf(x1 * c - x2 * sn);
    k_r[(size_t)row * DH_ + j + 64] = f2bf(x2 * c + x1 * sn);
  }
  if (t < DH_) {
    v_r[(size_t)row * DH_ + t] = f2bf(qkv[(size_t)row * NQKV + QS_ + DH_ + t]);
  }
}

// ---------------------------------------------------------------------------
// Flash attention (MQA): grid (B*NH, S/128). 8 waves x 16 q-rows each.
// Register-pipelined K/V staging; DPP-based softmax reductions.
// ---------------------------------------------------------------------------
__global__ __launch_bounds__(256) void attn_kernel(
    const unsigned short* __restrict__ q_r, const unsigned short* __restrict__ k_r,
    const unsigned short* __restrict__ v_r, unsigned short* __restrict__ o_out) {
  __shared__ unsigned short sQ[128 * 132];
  __shared__ unsigned short sK[32 * 132];
  __shared__ unsigned short sVT[128 * 36];
  __shared__ unsigned short sP[8 * 16 * 36];
  int bh = blockIdx.x, qb = blockIdx.y;
  int b = bh >> 5, h = bh & 31;
  int t = threadIdx.x, lane = t & 31, wid = t >> 5;
  int hi = lane >> 4, lo = lane & 15;
  const float scale = 0.08838834764f; // 1/sqrt(128)

  // stage Q (128x128 bf16)
  {
    int row = t >> 1, c0 = (t & 1) << 6;
    const uint2* src = (const uint2*)(q_r + ((size_t)bh * 1024 + qb * 128 + row) * DH_ + c0);
    uint2* dst = (uint2*)(sQ + row * 132 + c0);
#pragma unroll
    for (int j = 0; j < 16; ++j) dst[j] = src[j];
  }
  __syncthreads();

  v16bf aQ[4];
#pragma unroll
  for (int ks = 0; ks < 4; ++ks) aQ[ks] = load_frag(sQ + ks * 32, 132, wid * 16, lane);

  v8f O[8];
  float Mrow[8], Lrow[8];
#pragma unroll
  for (int dt = 0; dt < 8; ++dt)
#pragma unroll
    for (int r = 0; r < 8; ++r) O[dt][r] = 0.f;
#pragma unroll
  for (int r = 0; r < 8; ++r) { Mrow[r] = -1e30f; Lrow[r] = 0.f; }

  int krow = t >> 3, kc0 = (t & 7) << 4;       // K: 32 rows x 128 halves
  int vrow = (t >> 4) << 1, vc0 = (t & 15) << 3; // V: rows vrow,vrow+1, cols vc0..+7

  uint2 kreg[4];
  uint4 vreg0, vreg1;
  auto loadKV = [&](int kb) {
    const uint2* ks_ = (const uint2*)(k_r + ((size_t)(b * 1024 + kb * 32 + krow)) * DH_ + kc0);
    kreg[0] = ks_[0]; kreg[1] = ks_[1]; kreg[2] = ks_[2]; kreg[3] = ks_[3];
    const unsigned short* vsrc = v_r + ((size_t)(b * 1024 + kb * 32 + vrow)) * DH_ + vc0;
    vreg0 = *(const uint4*)vsrc;
    vreg1 = *(const uint4*)(vsrc + DH_);
  };

  int kbmax = qb * 4 + 4;
  loadKV(0);
  for (int kb = 0; kb < kbmax; ++kb) {
    __syncthreads();
    // commit staged K (row-major) and V (transposed, bf16 pairs packed)
    {
      uint2* kd = (uint2*)(sK + krow * 132 + kc0);
      kd[0] = kreg[0]; kd[1] = kreg[1]; kd[2] = kreg[2]; kd[3] = kreg[3];
      union { uint4 q; unsigned short s[8]; } ua, ub;
      ua.q = vreg0; ub.q = vreg1;
#pragma unroll
      for (int j = 0; j < 8; ++j) {
        *(unsigned int*)(sVT + (vc0 + j) * 36 + vrow) =
            (unsigned int)ua.s[j] | ((unsigned int)ub.s[j] << 16);
      }
    }
    __syncthreads();
    if (kb + 1 < kbmax) loadKV(kb + 1);

    // S = Q K^T : two 16x16 tiles (kv cols 0-15, 16-31)
    v8f s0, s1;
#pragma unroll
    for (int r = 0; r < 8; ++r) { s0[r] = 0.f; s1[r] = 0.f; }
#pragma unroll
    for (int ks = 0; ks < 4; ++ks) {
      v16bf b0 = load_frag(sK + ks * 32, 132, 0, lane);
      v16bf b1 = load_frag(sK + ks * 32, 132, 16, lane);
      s0 = wmma_bf16(aQ[ks], b0, s0);
      s1 = wmma_bf16(aQ[ks], b1, s1);
    }

    // online softmax (DPP reductions), write P tile to per-wave LDS scratch
    unsigned short* Pw = sP + wid * 16 * 36;
#pragma unroll
    for (int r = 0; r < 8; ++r) {
      int m_l = r + 8 * hi;
      int q_abs = qb * 128 + wid * 16 + m_l;
      int kv0 = kb * 32 + lo, kv1 = kv0 + 16;
      float v0 = (kv0 <= q_abs) ? s0[r] * scale : -1e30f;
      float v1 = (kv1 <= q_abs) ? s1[r] * scale : -1e30f;
      float mx = red_max16(fmaxf(v0, v1));
      float mnew = fmaxf(Mrow[r], mx);
      float alpha = __expf(Mrow[r] - mnew);
      float p0 = __expf(v0 - mnew), p1 = __expf(v1 - mnew);
      float rs = red_sum16(p0 + p1);
      Lrow[r] = Lrow[r] * alpha + rs;
      Mrow[r] = mnew;
#pragma unroll
      for (int dt = 0; dt < 8; ++dt) O[dt][r] *= alpha;
      Pw[m_l * 36 + lo]      = f2bf(p0);
      Pw[m_l * 36 + 16 + lo] = f2bf(p1);
    }

    // O += P(16x32) @ V(32x128)
    v16bf aP = load_frag(Pw, 36, 0, lane);
#pragma unroll
    for (int dt = 0; dt < 8; ++dt) {
      v16bf bv = load_frag(sVT, 36, dt * 16, lane);
      O[dt] = wmma_bf16(aP, bv, O[dt]);
    }
  }

  // epilogue: normalize and store in [b, s, h*128+d] bf16
#pragma unroll
  for (int r = 0; r < 8; ++r) {
    int m_l = r + 8 * hi;
    int row_g = b * 1024 + qb * 128 + wid * 16 + m_l;
    float linv = 1.f / Lrow[r];
#pragma unroll
    for (int dt = 0; dt < 8; ++dt) {
      int d = dt * 16 + lo;
      o_out[(size_t)row_g * HID + h * DH_ + d] = f2bf(O[dt][r] * linv);
    }
  }
}

// ---------------------------------------------------------------------------
// Router: softmax top-2 over 8 experts, slot assignment per expert.
// ---------------------------------------------------------------------------
__global__ __launch_bounds__(256) void route_kernel(
    const unsigned short* __restrict__ x, const float* __restrict__ gw,
    int* __restrict__ cnt, int* __restrict__ tok, int* __restrict__ krank,
    float* __restrict__ wt) {
  int wid = threadIdx.x >> 5, lane = threadIdx.x & 31;
  int tkn = blockIdx.x * 8 + wid;
  float p[8];
#pragma unroll
  for (int e = 0; e < 8; ++e) p[e] = 0.f;
  for (int i = lane; i < HID; i += 32) {
    float xv = bf2f(x[(size_t)tkn * HID + i]);
    const float4* g4 = (const float4*)(gw + (size_t)i * 8);
    float4 a0 = g4[0], a1 = g4[1];
    p[0] += xv * a0.x; p[1] += xv * a0.y; p[2] += xv * a0.z; p[3] += xv * a0.w;
    p[4] += xv * a1.x; p[5] += xv * a1.y; p[6] += xv * a1.z; p[7] += xv * a1.w;
  }
#pragma unroll
  for (int e = 0; e < 8; ++e)
#pragma unroll
    for (int m = 16; m >= 1; m >>= 1) p[e] += __shfl_xor(p[e], m, 32);
  if (lane == 0) {
    int e0 = 0;
#pragma unroll
    for (int e = 1; e < 8; ++e) if (p[e] > p[e0]) e0 = e;
    int e1 = (e0 == 0) ? 1 : 0;
#pragma unroll
    for (int e = 0; e < 8; ++e) if (e != e0 && p[e] > p[e1]) e1 = e;
    float w0 = 1.f / (1.f + __expf(p[e1] - p[e0]));
    float w1 = 1.f - w0;
    int s0 = atomicAdd(&cnt[e0], 1);
    tok[e0 * TOK + s0] = tkn; krank[e0 * TOK + s0] = 0; wt[e0 * TOK + s0] = w0;
    int s1 = atomicAdd(&cnt[e1], 1);
    tok[e1 * TOK + s1] = tkn; krank[e1 * TOK + s1] = 1; wt[e1 * TOK + s1] = w1;
  }
}

__global__ void zero_cnt_kernel(int* cnt) {
  if (threadIdx.x < 8) cnt[threadIdx.x] = 0;
}

// ---------------------------------------------------------------------------
// silu(g) * u ; optionally count-bounded per expert (gridDim.y = expert)
// ---------------------------------------------------------------------------
__global__ __launch_bounds__(256) void silu_mul_kernel(
    const unsigned short* __restrict__ gu, unsigned short* __restrict__ outp,
    int half, const int* __restrict__ cnt) {
  int row = blockIdx.x, e = blockIdx.y;
  const unsigned short* src;
  unsigned short* dst;
  if (cnt) {
    if (row >= cnt[e]) return;
    src = gu + (size_t)e * TOK * 2 * half + (size_t)row * 2 * half;
    dst = outp + (size_t)e * TOK * half + (size_t)row * half;
  } else {
    src = gu + (size_t)row * 2 * half;
    dst = outp + (size_t)row * half;
  }
  for (int i = threadIdx.x; i < half; i += 256) {
    float g = bf2f(src[i]), u = bf2f(src[i + half]);
    float s = g / (1.f + __expf(-g));
    dst[i] = f2bf(s * u);
  }
}

// ---------------------------------------------------------------------------
// Host launch
// ---------------------------------------------------------------------------
extern "C" void kernel_launch(void* const* d_in, const int* in_sizes, int n_in,
                              void* d_out, int out_size, void* d_ws, size_t ws_size,
                              hipStream_t stream) {
  const long long* positions = (const long long*)d_in[0];
  const float* hidden   = (const float*)d_in[1];
  const float* residual = (const float*)d_in[2];
  const float* qkv_w    = (const float*)d_in[3];
  const float* inorm_w  = (const float*)d_in[4];
  const float* wq_w     = (const float*)d_in[5];
  const float* o_w      = (const float*)d_in[6];
  const float* in_ln_w  = (const float*)d_in[7];
  const float* post_ln_w= (const float*)d_in[8];
  const float* gate_w   = (const float*)d_in[9];
  const float* w13      = (const float*)d_in[10];
  const float* w2       = (const float*)d_in[11];
  const float* sh_w13   = (const float*)d_in[12];
  const float* sh_w2    = (const float*)d_in[13];
  float* out = (float*)d_out;
  float* out_res2 = out + (size_t)TOK * HID;

  char* ws = (char*)d_ws;
  size_t off = 0;
  auto alloc = [&](size_t bytes) {
    size_t p = off;
    off = (off + bytes + 255) & ~(size_t)255;
    return p;
  };
  float*          res     = (float*)(ws + alloc((size_t)TOK * HID * 4));
  float*          qkv     = (float*)(ws + alloc((size_t)TOK * NQKV * 4));
  unsigned short* hb      = (unsigned short*)(ws + alloc((size_t)TOK * HID * 2));
  unsigned short* qn      = (unsigned short*)(ws + alloc((size_t)TOK * QS_ * 2));
  float*          qproj   = (float*)(ws + alloc((size_t)TOK * HID * 4));
  unsigned short* q_r     = (unsigned short*)(ws + alloc((size_t)TOK * HID * 2));
  unsigned short* k_r     = (unsigned short*)(ws + alloc((size_t)TOK * DH_ * 2));
  unsigned short* v_r     = (unsigned short*)(ws + alloc((size_t)TOK * DH_ * 2));
  unsigned short* attn    = (unsigned short*)(ws + alloc((size_t)TOK * HID * 2));
  unsigned short* h2      = (unsigned short*)(ws + alloc((size_t)TOK * HID * 2));
  unsigned short* gu_s    = (unsigned short*)(ws + alloc((size_t)TOK * HID * 2));
  unsigned short* a_s     = (unsigned short*)(ws + alloc((size_t)TOK * QS_ * 2));
  float*          out2    = (float*)(ws + alloc((size_t)TOK * 2 * HID * 4));
  unsigned short* gu_e    = (unsigned short*)(ws + alloc((size_t)E_ * TOK * 2 * MI_ * 2));
  unsigned short* eh      = (unsigned short*)(ws + alloc((size_t)E_ * TOK * MI_ * 2));
  int*            cnt     = (int*)(ws + alloc(E_ * 4));
  int*            tok     = (int*)(ws + alloc((size_t)E_ * TOK * 4));
  int*            krank   = (int*)(ws + alloc((size_t)E_ * TOK * 4));
  float*          wt      = (float*)(ws + alloc((size_t)E_ * TOK * 4));

  zero_cnt_kernel<<<1, 32, 0, stream>>>(cnt);

  // 1. res = hidden + residual ; h = rmsnorm(res, input_ln_w)
  add_rmsnorm_kernel<<<TOK, 256, 0, stream>>>(hidden, HID, residual, in_ln_w, res, hb, HID);

  GemmP p{};
  // 2. qkv = h @ qkv_w
  p.A = hb; p.B = qkv_w; p.M = TOK; p.N = NQKV; p.K = HID; p.lda = HID; p.ldc = NQKV;
  p.out_f32 = qkv;
  gemm_bf16_kernel<0, 0><<<dim3(NQKV / 128, TOK / 128), 256, 0, stream>>>(p);

  // 3. qn = rmsnorm(qkv[:, :2048], inter_norm_w)
  add_rmsnorm_kernel<<<TOK, 256, 0, stream>>>(qkv, NQKV, nullptr, inorm_w, nullptr, qn, QS_);

  // 4. qproj = qn @ wq_w
  p = GemmP{};
  p.A = qn; p.B = wq_w; p.M = TOK; p.N = HID; p.K = QS_; p.lda = QS_; p.ldc = HID;
  p.out_f32 = qproj;
  gemm_bf16_kernel<0, 0><<<dim3(HID / 128, TOK / 128), 256, 0, stream>>>(p);

  // 5. RoPE + repack q/k/v
  rope_kernel<<<TOK, 256, 0, stream>>>(positions, qproj, qkv, q_r, k_r, v_r);

  // 6. attention
  attn_kernel<<<dim3(64, 8), 256, 0, stream>>>(q_r, k_r, v_r, attn);

  // 7. res2 = attn @ o_w + res  -> d_out second tensor
  p = GemmP{};
  p.A = attn; p.B = o_w; p.M = TOK; p.N = HID; p.K = HID; p.lda = HID; p.ldc = HID;
  p.out_f32 = out_res2; p.res = res;
  gemm_bf16_kernel<0, 2><<<dim3(HID / 128, TOK / 128), 256, 0, stream>>>(p);

  // 8. h2 = rmsnorm(res2, post_ln_w)
  add_rmsnorm_kernel<<<TOK, 256, 0, stream>>>(out_res2, HID, nullptr, post_ln_w, nullptr, h2, HID);

  // 9. routing
  route_kernel<<<TOK / 8, 256, 0, stream>>>(h2, gate_w, cnt, tok, krank, wt);

  // 10. MoE expert GEMMs (top-2 gathered)
  for (int e = 0; e < E_; ++e) {
    p = GemmP{};
    p.A = h2; p.B = w13 + (size_t)e * HID * 2 * MI_;
    p.M = TOK; p.N = 2 * MI_; p.K = HID; p.lda = HID; p.ldc = 2 * MI_;
    p.out_b16 = gu_e + (size_t)e * TOK * 2 * MI_;
    p.rowidx = tok + e * TOK; p.cnt = cnt + e;
    gemm_bf16_kernel<1, 1><<<dim3((2 * MI_) / 128, TOK / 128), 256, 0, stream>>>(p);
  }
  silu_mul_kernel<<<dim3(TOK, E_), 256, 0, stream>>>(gu_e, eh, MI_, cnt);
  for (int e = 0; e < E_; ++e) {
    p = GemmP{};
    p.A = eh + (size_t)e * TOK * MI_; p.B = w2 + (size_t)e * MI_ * HID;
    p.M = TOK; p.N = HID; p.K = MI_; p.lda = MI_; p.ldc = HID;
    p.cnt = cnt + e; p.tok = tok + e * TOK; p.krank = krank + e * TOK;
    p.wt = wt + e * TOK; p.out2 = out2;
    gemm_bf16_kernel<2, 3><<<dim3(HID / 128, TOK / 128), 256, 0, stream>>>(p);
  }

  // 11. shared MLP
  p = GemmP{};
  p.A = h2; p.B = sh_w13; p.M = TOK; p.N = HID; p.K = HID; p.lda = HID; p.ldc = HID;
  p.out_b16 = gu_s;
  gemm_bf16_kernel<0, 1><<<dim3(HID / 128, TOK / 128), 256, 0, stream>>>(p);
  silu_mul_kernel<<<dim3(TOK, 1), 256, 0, stream>>>(gu_s, a_s, QS_, nullptr);

  // 12. d_out[0] = a_s @ share_w2 + out2[t][0] + out2[t][1]
  p = GemmP{};
  p.A = a_s; p.B = sh_w2; p.M = TOK; p.N = HID; p.K = QS_; p.lda = QS_; p.ldc = HID;
  p.out_f32 = out; p.out2 = out2;
  gemm_bf16_kernel<0, 4><<<dim3(HID / 128, TOK / 128), 256, 0, stream>>>(p);
}
